// CausalCTM_38388417691794
// MI455X (gfx1250) — matrix operand order repaired and checked by reading, not compiled, not run on hardware
//
// CausalCTM forward for MI455X (gfx1250, wave32, WMMA).
// Heavy GEMMs -> v_wmma_f32_16x16x32_bf16 (fp32 accum). A/B panels staged in
// LDS *in fragment order* (each lane's 16 bf16 contiguous -> ds_load_b128
// fragment reads, no half-packing movs). Global loads vectorized (b128) when
// strides permit. Each wave computes four 16x16 N-subtiles from one shared A
// fragment (4 WMMAs per k-step). Logits GEMM writes directly into d_out with
// (B,S,V,TICKS) strides; out_w (64 MB) stays resident in the 192 MB L2.
#include <hip/hip_runtime.h>
#include <math.h>

#define VCAB 32000
#define NN   512     // neurons
#define MMT  16      // trace length
#define TICKSC 4
#define DD   512
#define NSAC 256
#define NSOC 512
#define HHC  8
#define HDC  64
#define BBC  2
#define SSC  256
#define BSC  (BBC*SSC)   // 512

#define TN_PANEL 256     // N columns per block panel (4 waves x 4 subtiles x 16)

typedef __attribute__((ext_vector_type(16))) __bf16 v16bf;
typedef __attribute__((ext_vector_type(8)))  float  v8f;
typedef __attribute__((ext_vector_type(4)))  float  v4f;

// ---------------------------------------------------------------------------
// Generic strided, batched WMMA GEMM:  C = alpha * A @ B (+ bias[col])
// A[z, r, k], B[z, k, n], C[z, r, n]; z = (zb, zh) = (z/Hdim, z%Hdim).
// Block = 128 threads = 4 waves covering a 16 x 256 C panel.
//
// LDS fragment-order layouts (bf16, each lane's 16 values contiguous):
//   A:  lane = m + (k&16), slot = k&15
//   B:  sub = n>>4, lane = (n&15) | (k&16), slot = k&15
// so each WMMA operand is a single 32-byte aligned v16bf LDS load.
// OOB panel columns are zero-filled so WMMAs run unguarded (EXEC all-ones).
// ---------------------------------------------------------------------------
__global__ void __launch_bounds__(128)
wmma_gemm_kernel(const float* __restrict__ A,
                 long long sAb, long long sAh, long long sAr, long long sAk,
                 const float* __restrict__ Bm,
                 long long sBb, long long sBh, long long sBk, long long sBn,
                 float* __restrict__ C,
                 long long sCb, long long sCh, long long sCr, long long sCn,
                 const float* __restrict__ bias,
                 float alpha, int Mrows, int Ncols, int K, int Hdim)
{
    __shared__ __align__(32) __bf16 Asf[32 * 16];            // 1 KB
    __shared__ __align__(32) __bf16 Bsf[16 * 32 * 16];       // 16 KB

    const int tid  = threadIdx.x;
    const int wave = tid >> 5;
    const int lane = tid & 31;
    const int hi   = lane >> 4;
    const int l15  = lane & 15;

    const int tileM = blockIdx.y;
    const int nbase = blockIdx.x * TN_PANEL;
    const int z  = blockIdx.z;
    const int zb = z / Hdim;
    const int zh = z % Hdim;

    const long long baseA = (long long)zb * sAb + (long long)zh * sAh
                          + (long long)(tileM * 16) * sAr;
    const long long baseB = (long long)zb * sBb + (long long)zh * sBh;
    const long long baseC = (long long)zb * sCb + (long long)zh * sCh;

    const bool fastA = (sAk == 1);
    const bool fastB = (sBn == 1) && (nbase + TN_PANEL <= Ncols);

    v8f acc[4];
    #pragma unroll
    for (int s = 0; s < 4; ++s) acc[s] = v8f{0.f,0.f,0.f,0.f,0.f,0.f,0.f,0.f};

    for (int k0 = 0; k0 < K; k0 += 32) {
        // ---- A panel 16x32 -> LDS (fragment order) ----
        if (fastA) {
            // one float4 per thread (512 elements / 128 threads)
            int r  = tid >> 3;
            int cv = (tid & 7) << 2;
            v4f q = *(const v4f*)(A + baseA + (long long)r * sAr + (k0 + cv));
            #pragma unroll
            for (int e = 0; e < 4; ++e) {
                int k = cv + e;
                Asf[(r + (k & 16)) * 16 + (k & 15)] = (__bf16)q[e];
            }
        } else {
            for (int i = tid; i < 512; i += 128) {
                int r = i >> 5, k = i & 31;
                float v = A[baseA + (long long)r * sAr + (long long)(k0 + k) * sAk];
                Asf[(r + (k & 16)) * 16 + (k & 15)] = (__bf16)v;
            }
        }
        // ---- B panel 32x256 -> LDS (fragment order) ----
        if (fastB) {
            // 2048 float4s, 16 per thread, coalesced along n
            for (int i = tid; i < 2048; i += 128) {
                int r  = i >> 6;                 // k row within panel
                int cv = (i & 63) << 2;          // column within panel
                v4f q = *(const v4f*)(Bm + baseB + (long long)(k0 + r) * sBk
                                         + (nbase + cv));
                #pragma unroll
                for (int e = 0; e < 4; ++e) {
                    int c = cv + e;
                    Bsf[(c >> 4) * 512 + (((c & 15) | (r & 16)) * 16) + (r & 15)]
                        = (__bf16)q[e];
                }
            }
        } else {
            for (int i = tid; i < 32 * TN_PANEL; i += 128) {
                int r = i >> 8;
                int c = i & (TN_PANEL - 1);
                int gcol = nbase + c;
                float v = (gcol < Ncols)
                        ? Bm[baseB + (long long)(k0 + r) * sBk + (long long)gcol * sBn]
                        : 0.f;
                Bsf[(c >> 4) * 512 + (((c & 15) | (r & 16)) * 16) + (r & 15)] = (__bf16)v;
            }
        }
        __syncthreads();

        // one A fragment shared by this wave's 4 subtiles
        const v16bf af = *(const v16bf*)(&Asf[lane * 16]);
        #pragma unroll
        for (int sub = 0; sub < 4; ++sub) {
            const int subg = wave * 4 + sub;     // subtile index within panel
            const v16bf bf = *(const v16bf*)(&Bsf[subg * 512 + lane * 16]);
            acc[sub] = __builtin_amdgcn_wmma_f32_16x16x32_bf16(
                           false, af, false, bf, (short)0, acc[sub], false, false);
        }
        __syncthreads();
    }

    // ---- epilogue: C layout VGPR r -> M = r + 8*hi, N = lane&15 ----
    #pragma unroll
    for (int sub = 0; sub < 4; ++sub) {
        const int col = nbase + wave * 64 + sub * 16 + l15;
        if (col < Ncols) {                       // wave-uniform (Ncols % 16 == 0)
            const float bv = bias ? bias[col] : 0.f;
            #pragma unroll
            for (int r = 0; r < 8; ++r) {
                int row = tileM * 16 + hi * 8 + r;
                C[baseC + (long long)row * sCr + (long long)col * sCn]
                    = acc[sub][r] * alpha + bv;
            }
        }
    }
}

// ---------------------------------------------------------------------------
// Small fused kernels
// ---------------------------------------------------------------------------
__global__ void embed_kernel(const int* __restrict__ x, const float* __restrict__ tok,
                             const float* __restrict__ pos, float* __restrict__ emb)
{
    int bs = blockIdx.x;            // b*S + s
    int s = bs % SSC;
    long long tid = (long long)x[bs] * DD;
    for (int d = threadIdx.x; d < DD; d += blockDim.x)
        emb[(long long)bs * DD + d] = tok[tid + d] + pos[(long long)s * DD + d];
}

__global__ void __launch_bounds__(256)
ln_kernel(float* __restrict__ io, const float* __restrict__ g,
          const float* __restrict__ b, int L)
{
    __shared__ float red[256];
    long long base = (long long)blockIdx.x * L;
    float s = 0.f;
    for (int i = threadIdx.x; i < L; i += 256) s += io[base + i];
    red[threadIdx.x] = s; __syncthreads();
    for (int o = 128; o > 0; o >>= 1) {
        if (threadIdx.x < o) red[threadIdx.x] += red[threadIdx.x + o];
        __syncthreads();
    }
    float mean = red[0] / L;
    __syncthreads();
    float v = 0.f;
    for (int i = threadIdx.x; i < L; i += 256) { float d = io[base + i] - mean; v += d * d; }
    red[threadIdx.x] = v; __syncthreads();
    for (int o = 128; o > 0; o >>= 1) {
        if (threadIdx.x < o) red[threadIdx.x] += red[threadIdx.x + o];
        __syncthreads();
    }
    float rstd = rsqrtf(red[0] / L + 1e-5f);
    __syncthreads();
    for (int i = threadIdx.x; i < L; i += 256)
        io[base + i] = (io[base + i] - mean) * rstd * g[i] + b[i];
}

__global__ void init_state_kernel(const float* __restrict__ start_act,
                                  const float* __restrict__ start_trace,
                                  float* __restrict__ act, float* __restrict__ trace)
{
    long long g = (long long)blockIdx.x * blockDim.x + threadIdx.x;
    const long long actN = (long long)BSC * NN;
    if (g < actN) {
        long long bs = g / NN; int n = (int)(g % NN); int s = (int)(bs % SSC);
        act[g] = start_act[(long long)s * NN + n];
    }
    const long long trN = (long long)BSC * NN * MMT;
    if (g < trN) {
        long long bs = g / ((long long)NN * MMT);
        long long rem = g % ((long long)NN * MMT);
        int s = (int)(bs % SSC);
        trace[g] = start_trace[(long long)s * NN * MMT + rem];
    }
}

__global__ void sync_kernel(const float* __restrict__ act, const int* __restrict__ il,
                            const int* __restrict__ ir, const float* __restrict__ decay,
                            float* __restrict__ Aacc, float* __restrict__ Bacc,
                            float* __restrict__ syncv, int nsync, int first)
{
    long long g = (long long)blockIdx.x * blockDim.x + threadIdx.x;
    if (g >= (long long)BSC * nsync) return;
    long long bs = g / nsync; int i = (int)(g % nsync);
    float pp = act[bs * NN + il[i]] * act[bs * NN + ir[i]];
    float aa, bb;
    if (first) { aa = pp; bb = 1.f; }
    else {
        float r = expf(-decay[i]);
        aa = r * Aacc[g] + pp;
        bb = r * Bacc[g] + 1.f;
    }
    Aacc[g] = aa; Bacc[g] = bb;
    syncv[g] = aa / sqrtf(bb + 1e-8f);
}

__global__ void __launch_bounds__(256)
attn_softmax_kernel(float* __restrict__ sc)   // (B,H,S,S) in-place, causal mask
{
    __shared__ float red[256];
    int row = blockIdx.x;                      // b*H*S + h*S + q
    int q = row % SSC;
    long long base = (long long)row * SSC;
    int k = threadIdx.x;
    float v = sc[base + k];
    if (k > q) v = -1e9f;
    red[k] = v; __syncthreads();
    for (int o = 128; o > 0; o >>= 1) {
        if (k < o) red[k] = fmaxf(red[k], red[k + o]);
        __syncthreads();
    }
    float mx = red[0]; __syncthreads();
    float e = expf(v - mx);
    red[k] = e; __syncthreads();
    for (int o = 128; o > 0; o >>= 1) {
        if (k < o) red[k] += red[k + o];
        __syncthreads();
    }
    sc[base + k] = e / red[0];
}

__global__ void cat_kernel(const float* __restrict__ attn_out, const float* __restrict__ act,
                           float* __restrict__ cat)
{
    long long g = (long long)blockIdx.x * blockDim.x + threadIdx.x;
    if (g >= (long long)BSC * (DD + NN)) return;
    long long r = g / (DD + NN); int i = (int)(g % (DD + NN));
    cat[g] = (i < DD) ? attn_out[r * DD + i] : act[r * NN + (i - DD)];
}

__global__ void glu_kernel(const float* __restrict__ u, float* __restrict__ out)
{
    long long g = (long long)blockIdx.x * blockDim.x + threadIdx.x;
    if (g >= (long long)BSC * NN) return;
    long long r = g / NN; int i = (int)(g % NN);
    float a = u[r * 2 * NN + i];
    float b = u[r * 2 * NN + NN + i];
    out[g] = a * (1.f / (1.f + expf(-b)));
}

__global__ void trace_shift_kernel(float* __restrict__ trace, const float* __restrict__ state)
{
    long long g = (long long)blockIdx.x * blockDim.x + threadIdx.x;
    if (g >= (long long)BSC * NN) return;
    long long base = g * MMT;
    float tmp[MMT];
    #pragma unroll
    for (int m = 0; m < MMT - 1; ++m) tmp[m] = trace[base + m + 1];
    tmp[MMT - 1] = state[g];
    #pragma unroll
    for (int m = 0; m < MMT; ++m) trace[base + m] = tmp[m];
}

// Per-neuron NLM: trace(16) @ fc1_w[n](16x256) -> GLU -> 128 @ fc2_w[n](128x2)
// -> GLU -> act. One block per neuron, fc1_w[n] resident in LDS. K=16 is far
// too skinny for WMMA; this is VALU work by design.
__global__ void __launch_bounds__(256)
nlm_kernel(const float* __restrict__ trace, const float* __restrict__ fc1_w,
           const float* __restrict__ fc1_b, const float* __restrict__ fc2_w,
           const float* __restrict__ fc2_b, const float* __restrict__ nlm_T,
           float* __restrict__ act)
{
    __shared__ float w1[MMT * 256];
    __shared__ float b1[256];
    __shared__ float w2[256];
    __shared__ float b2s[2];
    __shared__ float tr[MMT];
    __shared__ float u[256];
    __shared__ float r0[128], r1[128];

    int n = blockIdx.x;
    int t = threadIdx.x;
    for (int i = t; i < MMT * 256; i += 256) w1[i] = fc1_w[(long long)n * MMT * 256 + i];
    b1[t] = fc1_b[(long long)n * 256 + t];
    if (t < 128) {
        w2[2 * t]     = fc2_w[(long long)n * 256 + 2 * t];
        w2[2 * t + 1] = fc2_w[(long long)n * 256 + 2 * t + 1];
    }
    if (t < 2) b2s[t] = fc2_b[n * 2 + t];
    float invT = 1.f / nlm_T[0];
    __syncthreads();

    for (int p = 0; p < BSC; ++p) {
        if (t < MMT) tr[t] = trace[((long long)p * NN + n) * MMT + t];
        __syncthreads();
        float s = b1[t];
        #pragma unroll
        for (int m = 0; m < MMT; ++m) s += tr[m] * w1[m * 256 + t];
        u[t] = s;
        __syncthreads();
        if (t < 128) {
            float hh = u[t] * (1.f / (1.f + expf(-u[t + 128])));
            r0[t] = hh * w2[2 * t];
            r1[t] = hh * w2[2 * t + 1];
        }
        __syncthreads();
        for (int o = 64; o > 0; o >>= 1) {
            if (t < o) { r0[t] += r0[t + o]; r1[t] += r1[t + o]; }
            __syncthreads();
        }
        if (t == 0) {
            float o0 = r0[0] + b2s[0], o1 = r1[0] + b2s[1];
            act[(long long)p * NN + n] = (o0 * (1.f / (1.f + expf(-o1)))) * invT;
        }
        __syncthreads();
    }
}

// Softmax entropy over V=32000 logits stored strided (stride TICKS) in d_out.
__global__ void __launch_bounds__(256)
entropy_kernel(const float* __restrict__ out, float* __restrict__ certs, int t)
{
    __shared__ float red[256];
    __shared__ float red2[256];
    int bs = blockIdx.x;
    const float* p0 = out + (long long)bs * VCAB * TICKSC + t;
    float mx = -1e30f;
    for (int v = threadIdx.x; v < VCAB; v += 256)
        mx = fmaxf(mx, p0[(long long)v * TICKSC]);
    red[threadIdx.x] = mx; __syncthreads();
    for (int o = 128; o > 0; o >>= 1) {
        if (threadIdx.x < o) red[threadIdx.x] = fmaxf(red[threadIdx.x], red[threadIdx.x + o]);
        __syncthreads();
    }
    mx = red[0]; __syncthreads();
    float se = 0.f, sxe = 0.f;
    for (int v = threadIdx.x; v < VCAB; v += 256) {
        float xv = p0[(long long)v * TICKSC] - mx;
        float e = expf(xv);
        se += e; sxe += e * xv;
    }
    red[threadIdx.x] = se; red2[threadIdx.x] = sxe; __syncthreads();
    for (int o = 128; o > 0; o >>= 1) {
        if (threadIdx.x < o) { red[threadIdx.x] += red[threadIdx.x + o];
                               red2[threadIdx.x] += red2[threadIdx.x + o]; }
        __syncthreads();
    }
    if (threadIdx.x == 0) {
        float S = red[0];
        float ent = (logf(S) - red2[0] / S) / logf((float)VCAB);
        certs[((long long)bs * 2 + 0) * TICKSC + t] = ent;
        certs[((long long)bs * 2 + 1) * TICKSC + t] = 1.f - ent;
    }
}

// ---------------------------------------------------------------------------
// Host orchestration
// ---------------------------------------------------------------------------
static void launch_gemm(hipStream_t st,
    const float* A, long long sAb, long long sAh, long long sAr, long long sAk,
    const float* Bm, long long sBb, long long sBh, long long sBk, long long sBn,
    float* C, long long sCb, long long sCh, long long sCr, long long sCn,
    const float* bias, float alpha, int M, int N, int K, int Z, int Hdim)
{
    dim3 grid((N + TN_PANEL - 1) / TN_PANEL, M / 16, Z);
    wmma_gemm_kernel<<<grid, 128, 0, st>>>(A, sAb, sAh, sAr, sAk,
                                           Bm, sBb, sBh, sBk, sBn,
                                           C, sCb, sCh, sCr, sCn,
                                           bias, alpha, M, N, K, Hdim);
}

static void gemm_plain(hipStream_t st, const float* A, const float* Bm,
                       const float* bias, float* C, int M, int N, int K)
{
    launch_gemm(st, A, 0, 0, K, 1, Bm, 0, 0, N, 1, C, 0, 0, N, 1,
                bias, 1.f, M, N, K, 1, 1);
}

extern "C" void kernel_launch(void* const* d_in, const int* in_sizes, int n_in,
                              void* d_out, int out_size, void* d_ws, size_t ws_size,
                              hipStream_t stream)
{
    (void)in_sizes; (void)n_in; (void)out_size; (void)ws_size;

    const int*   x        = (const int*)  d_in[0];
    const float* tok      = (const float*)d_in[1];
    const float* pos      = (const float*)d_in[2];
    const float* ln_emb_g = (const float*)d_in[3];
    const float* ln_emb_b = (const float*)d_in[4];
    const float* kv_w     = (const float*)d_in[5];
    const float* kv_b     = (const float*)d_in[6];
    const float* kv_ln_g  = (const float*)d_in[7];
    const float* kv_ln_b  = (const float*)d_in[8];
    const float* q_w      = (const float*)d_in[9];
    const float* q_b      = (const float*)d_in[10];
    const float* wq       = (const float*)d_in[11];
    const float* bq       = (const float*)d_in[12];
    const float* wk       = (const float*)d_in[13];
    const float* bk       = (const float*)d_in[14];
    const float* wv       = (const float*)d_in[15];
    const float* bv       = (const float*)d_in[16];
    const float* ow       = (const float*)d_in[17];
    const float* ob       = (const float*)d_in[18];
    const float* syn_w    = (const float*)d_in[19];
    const float* syn_b    = (const float*)d_in[20];
    const float* syn_g    = (const float*)d_in[21];
    const float* syn_beta = (const float*)d_in[22];
    const float* fc1_w    = (const float*)d_in[23];
    const float* fc1_b    = (const float*)d_in[24];
    const float* fc2_w    = (const float*)d_in[25];
    const float* fc2_b    = (const float*)d_in[26];
    const float* nlm_T    = (const float*)d_in[27];
    const float* start_act= (const float*)d_in[28];
    const float* start_tr = (const float*)d_in[29];
    const float* decay_a  = (const float*)d_in[30];
    const float* decay_o  = (const float*)d_in[31];
    const float* out_w    = (const float*)d_in[32];
    const float* out_b    = (const float*)d_in[33];
    const int* idx_al     = (const int*)d_in[34];
    const int* idx_ar     = (const int*)d_in[35];
    const int* idx_ol     = (const int*)d_in[36];
    const int* idx_or     = (const int*)d_in[37];

    float* preds = (float*)d_out;                                  // (B,S,V,TICKS)
    float* certs = (float*)d_out + (long long)BSC * VCAB * TICKSC; // (B,S,2,TICKS)

    // workspace carve-up (floats)
    float* ws = (float*)d_ws;
    size_t off = 0;
    auto alloc = [&](size_t n) { float* p = ws + off; off += n; return p; };
    float* emb    = alloc((size_t)BSC * DD);
    float* kv     = alloc((size_t)BSC * DD);
    float* kb     = alloc((size_t)BSC * DD);
    float* vb     = alloc((size_t)BSC * DD);
    float* act    = alloc((size_t)BSC * NN);
    float* trace  = alloc((size_t)BSC * NN * MMT);
    float* aacc   = alloc((size_t)BSC * NSAC);
    float* bacc   = alloc((size_t)BSC * NSAC);
    float* aocc   = alloc((size_t)BSC * NSOC);
    float* bocc   = alloc((size_t)BSC * NSOC);
    float* synca  = alloc((size_t)BSC * NSAC);
    float* synco  = alloc((size_t)BSC * NSOC);
    float* q1     = alloc((size_t)BSC * DD);
    float* qh     = alloc((size_t)BSC * DD);
    float* scores = alloc((size_t)BBC * HHC * SSC * SSC);
    float* attnf  = alloc((size_t)BSC * DD);
    float* attno  = alloc((size_t)BSC * DD);
    float* catb   = alloc((size_t)BSC * (DD + NN));
    float* ub     = alloc((size_t)BSC * 2 * NN);
    float* state  = alloc((size_t)BSC * NN);

    // ---- embedding + LN ----
    embed_kernel<<<BSC, 256, 0, stream>>>(x, tok, pos, emb);
    ln_kernel<<<BSC, 256, 0, stream>>>(emb, ln_emb_g, ln_emb_b, DD);

    // ---- kv = LN(emb @ kv_w + kv_b); k/v projections (fixed over ticks) ----
    gemm_plain(stream, emb, kv_w, kv_b, kv, BSC, DD, DD);
    ln_kernel<<<BSC, 256, 0, stream>>>(kv, kv_ln_g, kv_ln_b, DD);
    gemm_plain(stream, kv, wk, bk, kb, BSC, DD, DD);
    gemm_plain(stream, kv, wv, bv, vb, BSC, DD, DD);

    // ---- recurrent state init ----
    {
        long long trN = (long long)BSC * NN * MMT;
        init_state_kernel<<<(unsigned)((trN + 255) / 256), 256, 0, stream>>>(
            start_act, start_tr, act, trace);
    }

    const long long sKVb = (long long)SSC * HHC * HDC;   // 131072 per batch
    for (int t = 0; t < TICKSC; ++t) {
        // action synchronization
        sync_kernel<<<(BSC * NSAC + 255) / 256, 256, 0, stream>>>(
            act, idx_al, idx_ar, decay_a, aacc, bacc, synca, NSAC, t == 0);

        // q = (sync_a @ q_w + q_b) @ wq + bq
        gemm_plain(stream, synca, q_w, q_b, q1, BSC, DD, NSAC);
        gemm_plain(stream, q1, wq, bq, qh, BSC, DD, DD);

        // scores[b,h,q,k] = qh . k / sqrt(HD)   (Z = B*H batched)
        launch_gemm(stream,
            qh, sKVb, HDC, (long long)HHC * HDC, 1,
            kb, sKVb, HDC, 1, (long long)HHC * HDC,
            scores, (long long)HHC * SSC * SSC, (long long)SSC * SSC, SSC, 1,
            nullptr, 0.125f, SSC, SSC, HDC, BBC * HHC, HHC);

        attn_softmax_kernel<<<BBC * HHC * SSC, 256, 0, stream>>>(scores);

        // attn[b,q,h,d] = probs @ v
        launch_gemm(stream,
            scores, (long long)HHC * SSC * SSC, (long long)SSC * SSC, SSC, 1,
            vb, sKVb, HDC, (long long)HHC * HDC, 1,
            attnf, sKVb, HDC, (long long)HHC * HDC, 1,
            nullptr, 1.f, SSC, HDC, SSC, BBC * HHC, HHC);

        gemm_plain(stream, attnf, ow, ob, attno, BSC, DD, DD);

        // synapse: GLU(cat @ syn_w + syn_b) -> LN -> state
        cat_kernel<<<(BSC * (DD + NN) + 255) / 256, 256, 0, stream>>>(attno, act, catb);
        gemm_plain(stream, catb, syn_w, syn_b, ub, BSC, 2 * NN, DD + NN);
        glu_kernel<<<(BSC * NN + 255) / 256, 256, 0, stream>>>(ub, state);
        ln_kernel<<<BSC, 256, 0, stream>>>(state, syn_g, syn_beta, NN);

        // trace shift + per-neuron NLM -> act
        trace_shift_kernel<<<(BSC * NN + 255) / 256, 256, 0, stream>>>(trace, state);
        nlm_kernel<<<NN, 256, 0, stream>>>(trace, fc1_w, fc1_b, fc2_w, fc2_b, nlm_T, act);

        // output synchronization
        sync_kernel<<<(BSC * NSOC + 255) / 256, 256, 0, stream>>>(
            act, idx_ol, idx_or, decay_o, aocc, bocc, synco, NSOC, t == 0);

        // logits straight into d_out with (B,S,V,TICKS) strides; out_w stays in L2
        launch_gemm(stream,
            synco, 0, 0, NSOC, 1,
            out_w, 0, 0, VCAB, 1,
            preds + t, 0, 0, (long long)VCAB * TICKSC, TICKSC,
            out_b, 1.f, BSC, VCAB, NSOC, 1, 1);

        entropy_kernel<<<BSC, 256, 0, stream>>>((const float*)d_out, certs, t);
    }
}
